// HierarchicalUpsampleIGAModule_74191265071610
// MI455X (gfx1250) — matrix operand order repaired and checked by reading, not compiled, not run on hardware
//
#include <hip/hip_runtime.h>
#include <hip/hip_bf16.h>
#include <math.h>

// Problem constants (fixed by setup_inputs)
#define BB   2
#define KP   1024
#define MM   8
#define CC   256
#define KC   (KP * MM)          // 8192
#define ROWS_GATE  (BB * KC)    // 16384
#define ROWS_PRIOR (BB * KP)    // 2048

typedef __attribute__((ext_vector_type(16))) __bf16 v16bf;
typedef __attribute__((ext_vector_type(4)))  __bf16 v4bf;
typedef __attribute__((ext_vector_type(8)))  float  v8f;
typedef int v4i_vec __attribute__((vector_size(16)));

#if __has_builtin(__builtin_amdgcn_global_load_async_to_lds_b128)
#define ASYNC_COPY 1
typedef __attribute__((address_space(1))) v4i_vec GV4;
typedef __attribute__((address_space(3))) v4i_vec LV4;
#endif

__device__ __forceinline__ float wave_sum(float v) {
#pragma unroll
  for (int m = 16; m > 0; m >>= 1) v += __shfl_xor(v, m, 32);
  return v;
}

// score = -0.5*(||L^-1 d||^2 + logdet), L packed [l00,l10,l11,l20,l21,l22]
__device__ __forceinline__ float score3(float d0, float d1, float d2,
                                        const float* __restrict__ L, float ld) {
  float y0 = d0 / L[0];
  float y1 = (d1 - L[1] * y0) / L[2];
  float y2 = (d2 - L[3] * y0 - L[4] * y1) / L[5];
  return -0.5f * (y0 * y0 + y1 * y1 + y2 * y2 + ld);
}

// ---------------- Cholesky of Sigma_p + 1e-6 I, plus logdet ----------------
__global__ void chol_kernel(const float* __restrict__ Sigma,
                            float* __restrict__ Lp, float* __restrict__ logdet) {
  int idx = blockIdx.x * blockDim.x + threadIdx.x;
  if (idx >= ROWS_PRIOR) return;
  const float* S = Sigma + idx * 9;
  float s00 = S[0] + 1e-6f, s10 = S[3], s11 = S[4] + 1e-6f;
  float s20 = S[6], s21 = S[7], s22 = S[8] + 1e-6f;
  float l00 = sqrtf(s00);
  float l10 = s10 / l00;
  float l11 = sqrtf(s11 - l10 * l10);
  float l20 = s20 / l00;
  float l21 = (s21 - l20 * l10) / l11;
  float l22 = sqrtf(s22 - l20 * l20 - l21 * l21);
  float* L = Lp + idx * 6;
  L[0] = l00; L[1] = l10; L[2] = l11; L[3] = l20; L[4] = l21; L[5] = l22;
  logdet[idx] = 2.f * (logf(l00) + logf(l11) + logf(l22));
}

// ---------------- geom[t][d] = emb[t] . geom_W[:,d] + geom_b[d] ------------
__global__ void geom_kernel(const float* __restrict__ emb,
                            const float* __restrict__ gW,
                            const float* __restrict__ gb,
                            float* __restrict__ geom) {
  int l = threadIdx.x;
  if (l < MM * 3) {
    int t = l / 3, d = l % 3;
    float s = gb[d];
    for (int c = 0; c < CC; ++c) s += emb[t * CC + c] * gW[c * 3 + d];
    geom[l] = s;
  }
}

// ---- pre-pack W1 (256x256 f32) into bf16 WMMA B-fragment order, once ----
// dst linear index: ((kb*16 + nt)*32 + lane)*16 + e  where lane = half*16+ncol,
// source kk = (e&8 ? 16 : 0) + half*8 + (e&7), nn = nt*16 + ncol.
__global__ void w1pack_kernel(const float* __restrict__ W1, __bf16* __restrict__ dst) {
  int q = blockIdx.x * blockDim.x + threadIdx.x;   // 16384 threads, 4 elems each
  int d0 = q * 4;
  int e0 = d0 & 15;
  int ln = (d0 >> 4) & 31;
  int nt = (d0 >> 9) & 15;
  int kb = d0 >> 13;
  int wh = ln >> 4, nc = ln & 15;
  int nn = nt * 16 + nc;
  v4bf out;
#pragma unroll
  for (int u = 0; u < 4; ++u) {
    int kk = ((e0 & 8) << 1) + wh * 8 + (e0 & 7) + u;
    out[u] = (__bf16)W1[((kb * 32 + kk) << 8) + nn];
  }
  *(v4bf*)&dst[d0] = out;
}

// ------- fused LN -> GEMM(256x256, bf16 WMMA) -> SiLU -> dot W2 head -------
// One block = 16 rows, 8 waves, each wave owns 2 N-tiles of 16 columns.
// A is staged in fragment layout by the LN phase; W1 arrives pre-packed in
// fragment order, so per-K-chunk staging is a contiguous 16KB global->LDS
// copy (async-to-LDS when the toolchain exposes it), double-buffered.
__global__ void __launch_bounds__(256)
head_kernel(const float* __restrict__ s_parent, const float* __restrict__ emb,
            const float* __restrict__ ln_g, const float* __restrict__ ln_b,
            const __bf16* __restrict__ W1bf, const float* __restrict__ b1,
            const float* __restrict__ W2, const float* __restrict__ b2,
            float* __restrict__ outv, int gate_mode) {
  // [kb][lane][e] : A fragments for all 8 K-chunks (8 KB)
  __shared__ __align__(32) __bf16 a_fr[8][32][16];
  // double-buffered B fragments for one K-chunk (2 x 16 KB)
  __shared__ __align__(32) __bf16 w_buf[2][8192];
  __shared__ float out16[16];

  const int tid  = threadIdx.x;
  const int wid  = tid >> 5;
  const int lane = tid & 31;
  if (tid < 16) out16[tid] = 0.f;

  // ---- LayerNorm: wave `wid` handles rows 2*wid, 2*wid+1; scatter to a_fr ----
#pragma unroll
  for (int rr = 0; rr < 2; ++rr) {
    int row  = wid * 2 + rr;
    int grow = blockIdx.x * 16 + row;
    const float* xptr;
    const float* eptr = nullptr;
    if (gate_mode) {
      int bb = grow >> 13, ii = grow & (KC - 1);
      int jj = ii >> 3, tt = ii & 7;
      xptr = s_parent + (bb * KP + jj) * CC;
      eptr = emb + tt * CC;
    } else {
      int bb = grow >> 10, kk = grow & (KP - 1);
      xptr = s_parent + (bb * KP + kk) * CC;
    }
    float xs[8];
    float sum = 0.f, sq = 0.f;
#pragma unroll
    for (int q = 0; q < 2; ++q) {
      int c0 = q * 128 + lane * 4;
      float4 xv = *(const float4*)&xptr[c0];
      if (eptr) {
        float4 ev = *(const float4*)&eptr[c0];
        xv.x += ev.x; xv.y += ev.y; xv.z += ev.z; xv.w += ev.w;
      }
      xs[q * 4 + 0] = xv.x; xs[q * 4 + 1] = xv.y;
      xs[q * 4 + 2] = xv.z; xs[q * 4 + 3] = xv.w;
      sum += xv.x + xv.y + xv.z + xv.w;
      sq  += xv.x * xv.x + xv.y * xv.y + xv.z * xv.z + xv.w * xv.w;
    }
    sum = wave_sum(sum); sq = wave_sum(sq);
    float mean = sum * (1.f / 256.f);
    float var  = sq * (1.f / 256.f) - mean * mean;
    float rstd = rsqrtf(var + 1e-5f);
#pragma unroll
    for (int q = 0; q < 2; ++q) {
      int c0 = q * 128 + lane * 4;
      float4 gv = *(const float4*)&ln_g[c0];
      float4 bv = *(const float4*)&ln_b[c0];
      float hv[4];
      hv[0] = (xs[q * 4 + 0] - mean) * rstd * gv.x + bv.x;
      hv[1] = (xs[q * 4 + 1] - mean) * rstd * gv.y + bv.y;
      hv[2] = (xs[q * 4 + 2] - mean) * rstd * gv.z + bv.z;
      hv[3] = (xs[q * 4 + 3] - mean) * rstd * gv.w + bv.w;
#pragma unroll
      for (int u = 0; u < 4; ++u) {
        int c = c0 + u;
        // A 16x32 layout: half=(c>>3)&1, e=(c&7)|(((c>>4)&1)<<3), kb=c>>5
        int kb   = c >> 5;
        int half = (c >> 3) & 1;
        int e    = (c & 7) | (((c >> 4) & 1) << 3);
        a_fr[kb][half * 16 + row][e] = (__bf16)hv[u];
      }
    }
  }

  // ---- GEMM: H(16x256) = A(16x256) @ W1(256x256), bf16 -> f32 acc ----
  const int half = lane >> 4;
  const int ncol = lane & 15;
  const int nt0 = wid * 2, nt1 = wid * 2 + 1;
  v8f acc0 = {}; v8f acc1 = {};

  // contiguous 16KB chunk copy: 1024 int4 lines, 4 per thread
#define STAGE_CHUNK(kb_, buf_)                                                  \
  do {                                                                          \
    const int4* _src = (const int4*)(W1bf + (size_t)(kb_) * 8192);              \
    int4* _dst = (int4*)&w_buf[(buf_)][0];                                      \
    _Pragma("unroll")                                                           \
    for (int _r = 0; _r < 4; ++_r) {                                            \
      STAGE_ONE(_src + _r * 256 + tid, _dst + _r * 256 + tid);                  \
    }                                                                           \
  } while (0)

#ifdef ASYNC_COPY
#define STAGE_ONE(s_, d_)                                                       \
  __builtin_amdgcn_global_load_async_to_lds_b128((GV4*)(s_), (LV4*)(d_), 0, 0)
#define STAGE_WAIT()                                                            \
  do {                                                                          \
    __builtin_amdgcn_s_wait_asynccnt(0);                                        \
  } while (0)
#else
#define STAGE_ONE(s_, d_) *(d_) = *(s_)
#define STAGE_WAIT() do { } while (0)
#endif

  STAGE_CHUNK(0, 0);
  for (int kb = 0; kb < 8; ++kb) {
    int cur = kb & 1;
    STAGE_WAIT();
    __syncthreads();              // buffer `cur` (and a_fr on kb==0) ready
    if (kb + 1 < 8) STAGE_CHUNK(kb + 1, cur ^ 1);
    if (kb + 2 < 8)               // emits global_prefetch_b8
      __builtin_prefetch(W1bf + (size_t)(kb + 2) * 8192 + tid * 32, 0, 3);

    v16bf afrag = *(const v16bf*)&a_fr[kb][lane][0];
    v16bf bf0   = *(const v16bf*)&w_buf[cur][(nt0 * 32 + lane) * 16];
    v16bf bf1   = *(const v16bf*)&w_buf[cur][(nt1 * 32 + lane) * 16];
    acc0 = __builtin_amdgcn_wmma_f32_16x16x32_bf16(false, afrag, false, bf0,
                                                   (short)0, acc0, false, false);
    acc1 = __builtin_amdgcn_wmma_f32_16x16x32_bf16(false, afrag, false, bf1,
                                                   (short)0, acc1, false, false);
  }

  // ---- epilogue: SiLU(h + b1) dot W2, reduce per row via LDS atomics ----
  {
    int col0 = nt0 * 16 + ncol, col1 = nt1 * 16 + ncol;
    float b10 = b1[col0], b11 = b1[col1];
    float w20 = W2[col0], w21 = W2[col1];
#pragma unroll
    for (int v = 0; v < 8; ++v) {
      int row = v + 8 * half;   // C/D layout: M = vgpr + 8*(lane>=16)
      float h0 = acc0[v] + b10;
      atomicAdd(&out16[row], (h0 / (1.f + expf(-h0))) * w20);
      float h1 = acc1[v] + b11;
      atomicAdd(&out16[row], (h1 / (1.f + expf(-h1))) * w21);
    }
  }
  __syncthreads();

  if (tid < 16) {
    float val = out16[tid] + b2[0];
    int grow = blockIdx.x * 16 + tid;
    if (gate_mode) outv[grow] = val;                                    // raw b
    else outv[grow] = (val > 20.f) ? val : log1pf(expf(val));           // softplus
  }
}

// -------- argmax_{k != j} overlap score; self is always top-1 neighbor -----
__global__ void topk_kernel(const float* __restrict__ mu_p,
                            const float* __restrict__ Lp,
                            const float* __restrict__ logdet,
                            int* __restrict__ neigh1) {
  int wid = threadIdx.x >> 5, lane = threadIdx.x & 31;
  int gj = blockIdx.x * 8 + wid;           // 0..2047
  int b = gj >> 10, j = gj & (KP - 1);
  const float* mu = mu_p + (b * KP + j) * 3;
  float m0 = mu[0], m1 = mu[1], m2 = mu[2];
  float best = -3.4e38f; int bi = 0;
  for (int k = lane; k < KP; k += 32) {
    if (k == j) continue;
    const float* muk = mu_p + (b * KP + k) * 3;
    const float* L   = Lp + (b * KP + k) * 6;
    float sc = score3(m0 - muk[0], m1 - muk[1], m2 - muk[2], L, logdet[b * KP + k]);
    if (sc > best) { best = sc; bi = k; }
  }
#pragma unroll
  for (int m = 16; m > 0; m >>= 1) {
    float ob = __shfl_xor(best, m, 32);
    int   oi = __shfl_xor(bi, m, 32);
    if (ob > best || (ob == best && oi < bi)) { best = ob; bi = oi; }
  }
  if (lane == 0) neigh1[gj] = bi;
}

// -------- per-child: score2/softmax, mixed mu/Sigma, gated channel mix -----
__global__ void __launch_bounds__(256)
child_kernel(const float* __restrict__ s_parent, const float* __restrict__ mu_p,
             const float* __restrict__ Sigma_p, const float* __restrict__ mask_parent,
             const float* __restrict__ xi_noise, const float* __restrict__ Lp,
             const float* __restrict__ logdet, const float* __restrict__ geom,
             const float* __restrict__ a_prior, const float* __restrict__ bgate,
             const int* __restrict__ neigh1,
             float* __restrict__ out_s, float* __restrict__ out_mu,
             float* __restrict__ out_Sig, float* __restrict__ out_g) {
  int wid = threadIdx.x >> 5, lane = threadIdx.x & 31;
  int idx = blockIdx.x * 8 + wid;          // 0..16383
  int b = idx >> 13, i = idx & (KC - 1);
  int j = i >> 3, t = i & 7;
  int pj = b * KP + j;

  // mu0 = mu_p[j] + L[j] @ (xi_noise + geom[t])
  const float* L  = Lp + pj * 6;
  const float* xn = xi_noise + (pj * MM + t) * 3;
  float x0 = xn[0] + geom[t * 3 + 0];
  float x1 = xn[1] + geom[t * 3 + 1];
  float x2 = xn[2] + geom[t * 3 + 2];
  const float* muj = mu_p + pj * 3;
  float m0 = muj[0] + L[0] * x0;
  float m1 = muj[1] + L[1] * x0 + L[2] * x1;
  float m2 = muj[2] + L[3] * x0 + L[4] * x1 + L[5] * x2;

  int n1 = neigh1[pj];
  int p0 = pj, p1 = b * KP + n1;
  const float* mu0p = mu_p + p0 * 3;
  const float* mu1p = mu_p + p1 * 3;
  float s0 = score3(m0 - mu0p[0], m1 - mu0p[1], m2 - mu0p[2], Lp + p0 * 6, logdet[p0]);
  float s1 = score3(m0 - mu1p[0], m1 - mu1p[1], m2 - mu1p[2], Lp + p1 * 6, logdet[p1]);
  float mx = fmaxf(s0, s1);
  float e0 = expf(s0 - mx), e1 = expf(s1 - mx);
  float inv = 1.f / (e0 + e1);
  float w0 = e0 * inv, w1 = e1 * inv;
  float mc0 = w0 * mu0p[0] + w1 * mu1p[0];
  float mc1 = w0 * mu0p[1] + w1 * mu1p[1];
  float mc2 = w0 * mu0p[2] + w1 * mu1p[2];

  // gate value: b + BETA*log(a + 1e-8) -> sigmoid * mask
  float av   = a_prior[pj];
  float bval = bgate[idx] + 0.5f * logf(av + 1e-8f);
  float g = (1.f / (1.f + expf(-bval))) * mask_parent[pj];

  // gated channel mix (256 wide, one wave, float4 streams)
  const float* sp0 = s_parent + (size_t)p0 * CC;
  const float* sp1 = s_parent + (size_t)p1 * CC;
  float* so = out_s + (size_t)idx * CC;
#pragma unroll
  for (int q = 0; q < 2; ++q) {
    int c = q * 128 + lane * 4;
    float4 v0 = *(const float4*)&sp0[c];
    float4 v1 = *(const float4*)&sp1[c];
    float4 o;
    o.x = g * (w0 * v0.x + w1 * v1.x);
    o.y = g * (w0 * v0.y + w1 * v1.y);
    o.z = g * (w0 * v0.z + w1 * v1.z);
    o.w = g * (w0 * v0.w + w1 * v1.w);
    *(float4*)&so[c] = o;
  }

  if (lane == 0) {
    float* om = out_mu + idx * 3;
    om[0] = mc0; om[1] = mc1; om[2] = mc2;
    const float* S0 = Sigma_p + p0 * 9;
    const float* S1 = Sigma_p + p1 * 9;
    float dm0[3] = { mu0p[0] - mc0, mu0p[1] - mc1, mu0p[2] - mc2 };
    float dm1[3] = { mu1p[0] - mc0, mu1p[1] - mc1, mu1p[2] - mc2 };
    float* oS = out_Sig + idx * 9;
    const float ip2 = 0.390625f;   // PHI^-2, PHI=1.6
#pragma unroll
    for (int a = 0; a < 3; ++a)
#pragma unroll
      for (int c2 = 0; c2 < 3; ++c2) {
        float v = (w0 * S0[a * 3 + c2] + w1 * S1[a * 3 + c2]) * ip2
                + w0 * dm0[a] * dm0[c2] + w1 * dm1[a] * dm1[c2];
        if (a == c2) v += 1e-4f;   // JITTER
        oS[a * 3 + c2] = v;
      }
    out_g[idx] = g;
  }
}

// ---------------- deterministic count loss over g ----------------
__global__ void loss_kernel(const float* __restrict__ g,
                            const int* __restrict__ Ktp,
                            float* __restrict__ out) {
  __shared__ float sh[256];
  float Kt = (float)Ktp[0];
  float total = 0.f;
  for (int b = 0; b < BB; ++b) {
    float s = 0.f;
    for (int p = threadIdx.x; p < KC; p += 256) s += g[b * KC + p];
    sh[threadIdx.x] = s;
    __syncthreads();
    for (int st = 128; st > 0; st >>= 1) {
      if (threadIdx.x < st) sh[threadIdx.x] += sh[threadIdx.x + st];
      __syncthreads();
    }
    if (threadIdx.x == 0) { float d = sh[0] - Kt; total += d * d; }
    __syncthreads();
  }
  if (threadIdx.x == 0) out[0] = total * (1.f / BB);
}

extern "C" void kernel_launch(void* const* d_in, const int* in_sizes, int n_in,
                              void* d_out, int out_size, void* d_ws, size_t ws_size,
                              hipStream_t stream) {
  const float* s_parent    = (const float*)d_in[0];
  const float* mu_p        = (const float*)d_in[1];
  const float* Sigma_p     = (const float*)d_in[2];
  const float* mask_parent = (const float*)d_in[3];
  const float* xi_noise    = (const float*)d_in[4];
  const float* emb         = (const float*)d_in[5];
  const float* gate_ln_g   = (const float*)d_in[6];
  const float* gate_ln_b   = (const float*)d_in[7];
  const float* gate_W1     = (const float*)d_in[8];
  const float* gate_b1     = (const float*)d_in[9];
  const float* gate_W2     = (const float*)d_in[10];
  const float* gate_b2     = (const float*)d_in[11];
  const float* prior_ln_g  = (const float*)d_in[12];
  const float* prior_ln_b  = (const float*)d_in[13];
  const float* prior_W1    = (const float*)d_in[14];
  const float* prior_b1    = (const float*)d_in[15];
  const float* prior_W2    = (const float*)d_in[16];
  const float* prior_b2    = (const float*)d_in[17];
  const float* geom_W      = (const float*)d_in[18];
  const float* geom_b      = (const float*)d_in[19];
  const int*   Ktp         = (const int*)d_in[20];

  // workspace layout (float offsets)
  float* ws      = (float*)d_ws;
  float*  Lp      = ws;                 // 2048*6 = 12288
  float*  logdet  = ws + 12288;         // 2048
  float*  geom    = ws + 14336;         // 24 (padded to 64)
  float*  a_prior = ws + 14400;         // 2048
  float*  bgate   = ws + 16448;         // 16384
  int*    neigh1  = (int*)(ws + 32832); // 2048 ints
  __bf16* w1g_bf  = (__bf16*)(ws + 34880); // 65536 bf16 (= 32768 floats)
  __bf16* w1p_bf  = (__bf16*)(ws + 67648); // 65536 bf16

  // output layout (floats, concatenated in return order)
  float* out_s    = (float*)d_out;                     // B*Kc*C = 4194304
  float* out_mu   = out_s  + (size_t)ROWS_GATE * CC;   // 49152
  float* out_Sig  = out_mu + (size_t)ROWS_GATE * 3;    // 147456
  float* out_g    = out_Sig + (size_t)ROWS_GATE * 9;   // 16384
  float* out_loss = out_g  + ROWS_GATE;                // 1

  chol_kernel<<<ROWS_PRIOR / 256, 256, 0, stream>>>(Sigma_p, Lp, logdet);
  geom_kernel<<<1, 32, 0, stream>>>(emb, geom_W, geom_b, geom);
  w1pack_kernel<<<64, 256, 0, stream>>>(gate_W1, w1g_bf);
  w1pack_kernel<<<64, 256, 0, stream>>>(prior_W1, w1p_bf);
  head_kernel<<<ROWS_PRIOR / 16, 256, 0, stream>>>(
      s_parent, nullptr, prior_ln_g, prior_ln_b, w1p_bf, prior_b1,
      prior_W2, prior_b2, a_prior, 0);
  head_kernel<<<ROWS_GATE / 16, 256, 0, stream>>>(
      s_parent, emb, gate_ln_g, gate_ln_b, w1g_bf, gate_b1,
      gate_W2, gate_b2, bgate, 1);
  topk_kernel<<<ROWS_PRIOR / 8, 256, 0, stream>>>(mu_p, Lp, logdet, neigh1);
  child_kernel<<<ROWS_GATE / 8, 256, 0, stream>>>(
      s_parent, mu_p, Sigma_p, mask_parent, xi_noise, Lp, logdet, geom,
      a_prior, bgate, neigh1, out_s, out_mu, out_Sig, out_g);
  loss_kernel<<<1, 256, 0, stream>>>(out_g, Ktp, out_loss);
}
